// LIFLayer_65017214926967
// MI455X (gfx1250) — matrix operand order repaired and checked by reading, not compiled, not run on hardware
//
#include <hip/hip_runtime.h>

// ---------------------------------------------------------------------------
// LIF layer for MI455X (gfx1250, wave32).
//   Phase 0: repack W (f32 512x512) -> bf16 in WMMA-B-friendly layout (512 KB ws)
//   Phase 1: GEMM i_input = spikes @ W + b  via v_wmma_f32_16x16x32_bf16,
//            written directly into d_out (no scratch round trip).
//   Phase 2: sequential LIF scan over T, in place on d_out.
// Roofline: GEMM ~128 MiB traffic @ 23.3 TB/s ~ 5.5us, WMMA flops negligible;
// scan ~128 MiB ~ 5.5us across 16384 independent chains (512 waves).
// ---------------------------------------------------------------------------

typedef __attribute__((ext_vector_type(16))) __bf16  v16bf;
typedef __attribute__((ext_vector_type(2)))  __bf16  bf16x2;
typedef __attribute__((ext_vector_type(8)))  float   v8f;
typedef __attribute__((ext_vector_type(2)))  float   f32x2;
typedef __attribute__((ext_vector_type(4)))  float   f32x4;
typedef __attribute__((ext_vector_type(4)))  unsigned int u32x4;

#define BATCH   32
#define TSTEPS  1024
#define DIM     512          // K
#define HID     512          // N
#define MTOT    (BATCH * TSTEPS)   // 32768

#define ALPHA_MEM 0.9512294245007140f   // exp(-0.001/0.020)
#define ALPHA_SYN 0.8187307530779818f   // exp(-0.001/0.005)

// Single-instruction packed conversion: 2 x f32 -> packed bf16x2 dword
// (lowers to v_cvt_pk_bf16_f32 dst, a, b).
static __device__ __forceinline__ unsigned int pack_bf16(float a, float b) {
    union { bf16x2 h; unsigned int u; } p;
    f32x2 f = {a, b};
    p.h = __builtin_convertvector(f, bf16x2);
    return p.u;
}

union Frag16 { u32x4 q[2]; v16bf v; };

// ---------------------------------------------------------------------------
// Phase 0: pack W[k][n] (f32, row-major K x N) into bf16 dwords arranged so a
// B-fragment (16x16x32 WMMA, K block of 32) is two contiguous b128 loads/lane.
// Layout (dword index): (((ntile*64 + kgroup)*16 + nlane)*4 + ksub)
//   ntile = n>>4, nlane = n&15, kpair = k>>1, kgroup = kpair>>2, ksub = kpair&3
// ---------------------------------------------------------------------------
__global__ void pack_w_kernel(const float* __restrict__ W,
                              unsigned int* __restrict__ Wpk) {
    int idx   = blockIdx.x * blockDim.x + threadIdx.x;  // 0 .. 131071
    int kpair = idx >> 9;                               // 0 .. 255
    int n     = idx & (HID - 1);                        // 0 .. 511
    float w0 = W[(2 * kpair)     * HID + n];
    float w1 = W[(2 * kpair + 1) * HID + n];
    int ntile = n >> 4, nlane = n & 15;
    int kg = kpair >> 2, ks = kpair & 3;
    Wpk[(((ntile * 64 + kg) * 16 + nlane) << 2) + ks] = pack_bf16(w0, w1);
}

// ---------------------------------------------------------------------------
// Phase 1: GEMM. Block = 256 threads = 8 waves (2 M-waves x 4 N-waves).
// Block tile 128(M) x 128(N); wave tile 64 x 32 (4x2 WMMA accumulators).
// K loop in steps of 32 (one 16x16x32 bf16 WMMA per tile pair per step).
// ---------------------------------------------------------------------------
__global__ void __launch_bounds__(256)
gemm_bias_kernel(const float* __restrict__ A,          // spikes [MTOT][DIM]
                 const unsigned int* __restrict__ Wpk, // packed W
                 const float* __restrict__ bias,       // [HID]
                 float* __restrict__ out) {            // [MTOT][HID]
    __shared__ unsigned int ldsA[128][16];   // 128 rows x 16 kpairs (bf16x2) = 8 KB

    const int tid   = threadIdx.x;
    const int lane  = tid & 31;
    const int wave  = tid >> 5;          // 0..7
    const int wm    = wave >> 2;         // 0..1  (M direction)
    const int wn    = wave & 3;          // 0..3  (N direction)
    const int lane16   = lane & 15;
    const int lanehalf = lane >> 4;      // 0/1 : K half for A/B fragments

    const int blockM = blockIdx.x;       // 0..255
    const int blockN = blockIdx.y;       // 0..3

    const int mWave = blockM * 128 + wm * 64;
    const int nWave = blockN * 128 + wn * 32;

    // Staging assignment: each thread owns one LDS row-half (16 f32 -> 8 dwords)
    const int srow  = tid >> 1;          // 0..127
    const int shalf = tid & 1;           // 0/1 : K sub-half of 16
    const float* aRow = A + (size_t)(blockM * 128 + srow) * DIM + shalf * 16;

    const v8f vzero = {0.f, 0.f, 0.f, 0.f, 0.f, 0.f, 0.f, 0.f};
    v8f acc[4][2];
#pragma unroll
    for (int mt = 0; mt < 4; ++mt)
#pragma unroll
        for (int nt = 0; nt < 2; ++nt) acc[mt][nt] = vzero;

    for (int kb = 0; kb < DIM; kb += 32) {
        // ---- stage A tile (128 x 32 f32 -> bf16 pairs) into LDS ----
        {
            const float* s = aRow + kb;
            f32x4 f0 = *(const f32x4*)(s + 0);
            f32x4 f1 = *(const f32x4*)(s + 4);
            f32x4 f2 = *(const f32x4*)(s + 8);
            f32x4 f3 = *(const f32x4*)(s + 12);
            u32x4 p0 = {pack_bf16(f0[0], f0[1]), pack_bf16(f0[2], f0[3]),
                        pack_bf16(f1[0], f1[1]), pack_bf16(f1[2], f1[3])};
            u32x4 p1 = {pack_bf16(f2[0], f2[1]), pack_bf16(f2[2], f2[3]),
                        pack_bf16(f3[0], f3[1]), pack_bf16(f3[2], f3[3])};
            *(u32x4*)&ldsA[srow][shalf * 8 + 0] = p0;   // ds_store_b128
            *(u32x4*)&ldsA[srow][shalf * 8 + 4] = p1;   // ds_store_b128
            // Prefetch next K tile into WGP-level caches (locality 3 -> WGP scope)
            if (kb + 32 < DIM) __builtin_prefetch(s + 32, 0, 3);
        }
        __syncthreads();

        // ---- B fragments for this K block (from L2-resident packed W) ----
        // B layout used here (contiguous k): lane: N = lane&15, K = (lane>>4)*16 + {2v, 2v+1}
        Frag16 bf[2];
        {
            const int kpair0 = (kb >> 1) + lanehalf * 8;
            const int kg0    = kpair0 >> 2;
#pragma unroll
            for (int nt = 0; nt < 2; ++nt) {
                const int ntile = (nWave >> 4) + nt;
                const unsigned int* p =
                    Wpk + ((((ntile * 64 + kg0) * 16) + lane16) << 2);
                bf[nt].q[0] = *(const u32x4*)(p);        // VGPRs 0..3 (K +0..7)
                bf[nt].q[1] = *(const u32x4*)(p + 64);   // VGPRs 4..7 (K +8..15 of half)
            }
        }

        // ---- A fragments + WMMA ----
        // A layout: lane: M = lane&15, VGPR0..3 = K lanehalf*16+{0..7},
        //               VGPR4..7 = K lanehalf*16+8+{0..7}
#pragma unroll
        for (int mt = 0; mt < 4; ++mt) {
            const int row = wm * 64 + mt * 16 + lane16;
            Frag16 af;
            af.q[0] = *(const u32x4*)&ldsA[row][lanehalf * 8];
            af.q[1] = *(const u32x4*)&ldsA[row][lanehalf * 8 + 4];
#pragma unroll
            for (int nt = 0; nt < 2; ++nt) {
                acc[mt][nt] = __builtin_amdgcn_wmma_f32_16x16x32_bf16(
                    false, af.v, false, bf[nt].v,
                    (short)0, acc[mt][nt], false, false);
            }
        }
        __syncthreads();
    }

    // ---- epilogue: add bias, store f32 ----
    // C layout: VGPR r: M = (lane>>4)*8 + r, N = lane&15 (per 16x16 tile)
    const float bias0 = bias[nWave + lane16];
    const float bias1 = bias[nWave + 16 + lane16];
#pragma unroll
    for (int mt = 0; mt < 4; ++mt) {
        const int mBase = mWave + mt * 16 + lanehalf * 8;
#pragma unroll
        for (int nt = 0; nt < 2; ++nt) {
            const int nCol = nWave + nt * 16 + lane16;
            const float bi = nt ? bias1 : bias0;
#pragma unroll
            for (int r = 0; r < 8; ++r)
                out[(size_t)(mBase + r) * HID + nCol] = acc[mt][nt][r] + bi;
        }
    }
}

// ---------------------------------------------------------------------------
// Phase 2: LIF scan over T, in place on d_out. 16384 independent chains.
// Lane owns one (batch, hidden) column: read i_input[t], write spike[t].
// ---------------------------------------------------------------------------
__global__ void __launch_bounds__(256)
lif_scan_kernel(float* __restrict__ io) {
    const int idx = blockIdx.x * blockDim.x + threadIdx.x;  // 0 .. 16383
    const int b   = idx >> 9;
    const int h   = idx & (HID - 1);
    float* p = io + (size_t)b * TSTEPS * HID + h;

    float v = 0.f, syn = 0.f;
#pragma unroll 4
    for (int t = 0; t < TSTEPS; ++t) {
        const float it = p[(size_t)t * HID];
        syn = fmaf(ALPHA_SYN, syn, it);
        v   = fmaf(ALPHA_MEM, v, syn);
        const bool fire = (v >= 1.0f);
        p[(size_t)t * HID] = fire ? 1.0f : 0.0f;
        v = fire ? 0.0f : v;
    }
}

// ---------------------------------------------------------------------------
extern "C" void kernel_launch(void* const* d_in, const int* in_sizes, int n_in,
                              void* d_out, int out_size, void* d_ws, size_t ws_size,
                              hipStream_t stream) {
    const float* spikes = (const float*)d_in[0];   // [32,1024,512]
    const float* W      = (const float*)d_in[1];   // [512,512]
    const float* b      = (const float*)d_in[2];   // [512]
    float* out          = (float*)d_out;           // [32,1024,512]
    unsigned int* Wpk   = (unsigned int*)d_ws;     // 512 KB packed bf16 W

    // Phase 0: repack W (131072 dwords)
    pack_w_kernel<<<dim3(512), dim3(256), 0, stream>>>(W, Wpk);

    // Phase 1: GEMM -> d_out  (grid: 32768/128 x 512/128)
    gemm_bias_kernel<<<dim3(MTOT / 128, HID / 128), dim3(256), 0, stream>>>(
        spikes, Wpk, b, out);

    // Phase 2: in-place LIF scan (16384 chains)
    lif_scan_kernel<<<dim3(16384 / 256), dim3(256), 0, stream>>>(out);
}
